// AttentionWithSpatialFeature_79645873537331
// MI455X (gfx1250) — matrix operand order repaired and checked by reading, not compiled, not run on hardware
//
#include <hip/hip_runtime.h>
#include <hip/hip_bf16.h>

// ---------------------------------------------------------------------------
// AttentionWithSpatialFeature for MI455X (gfx1250, wave32, WMMA 16x16x32 bf16)
//
// Pipeline (all on `stream`):
//   1. cvt fp32->bf16 : hidden, spatial; weights transposed [out,in]; rel table
//      padded to 272 rows.
//   2. k_proj x5      : bf16 WMMA GEMM  X@W+b -> Q,K,SQ,SK (layout [bh,S,64])
//                       and V transposed (layout [bh,64,S]).
//   3. k_relpos       : G[bh][m][r] = q_m . T_r  (bf16 WMMA GEMM) so that
//                       rel_bias[i,m] = G[m, clip(m-i)+128]  (gather, no VALU dots)
//   4. k_attn_fused   : per (bh, 16-row tile): scores (4 WMMA per 16x16 tile:
//                       qk + spatial, K=64 each) + scale + G-gather, kept in
//                       registers; cross-wave softmax via LDS stats; weights
//                       written to d_out ONCE; P staged bf16 in LDS; P@V via
//                       WMMA from LDS with ds_add_f32 cross-wave reduction.
// ---------------------------------------------------------------------------

typedef __bf16 bf16;
typedef __attribute__((ext_vector_type(16))) __bf16 v16bf;
typedef __attribute__((ext_vector_type(8)))  __bf16 v8bf;
typedef __attribute__((ext_vector_type(8)))  float  v8f;

#define B_      4
#define S_      1024
#define HID_    1024
#define NH_     16
#define HD_     64
#define BH_     (B_ * NH_)
#define MAXREL_ 128
#define RELROWS_ 257
#define RELPAD_  272
#define SCALE_  0.125f   // 1/sqrt(64)

union FragU { v16bf v; v8bf h[2]; };

__device__ __forceinline__ int lane_id() { return (int)(threadIdx.x & 31); }

// A operand, 16x32 bf16 (ISA 7.12.2): lanes 0-15 hold K = k0..k0+7 (VGPR0-3)
// and k0+16..k0+23 (VGPR4-7); lanes 16-31 hold k0+8.. and k0+24.. .
__device__ __forceinline__ v16bf load_a_frag(const bf16* base, int row, int ld, int k0) {
  int lane = lane_id(); int half = lane >> 4; int r = lane & 15;
  const bf16* p = base + (size_t)(row + r) * ld + k0 + half * 8;
  FragU f;
  f.h[0] = *reinterpret_cast<const v8bf*>(p);
  f.h[1] = *reinterpret_cast<const v8bf*>(p + 16);
  return f.v;
}

// B operand, 32x16 bf16: lane r holds column (col0+r) of B == row (col0+r) of
// the row-major source; lanes 0-15: K = k0..k0+7 and k0+16..k0+23, lanes 16-31:
// K = k0+8..k0+15 and k0+24..k0+31 (same K order as the A operand).
__device__ __forceinline__ v16bf load_b_frag(const bf16* base, int col0, int ld, int k0) {
  int lane = lane_id(); int half = lane >> 4; int r = lane & 15;
  const bf16* p = base + (size_t)(col0 + r) * ld + k0 + half * 8;
  FragU f;
  f.h[0] = *reinterpret_cast<const v8bf*>(p);
  f.h[1] = *reinterpret_cast<const v8bf*>(p + 16);
  return f.v;
}

__device__ __forceinline__ v8f wmma_bf16(v16bf a, v16bf b, v8f c) {
  return __builtin_amdgcn_wmma_f32_16x16x32_bf16(false, a, false, b,
                                                 (short)0, c, false, false);
}

// ---------------------------------------------------------------------------
// Conversion kernels
// ---------------------------------------------------------------------------
__global__ void k_cvt(const float* __restrict__ src, bf16* __restrict__ dst, int n) {
  int i = blockIdx.x * blockDim.x + threadIdx.x;
  if (i < n) dst[i] = (bf16)src[i];
}

// W stored [in=1024, out=1024]; produce Wt[out][in] so WMMA-B reads are rows.
__global__ void k_cvt_transpose(const float* __restrict__ W, bf16* __restrict__ Wt) {
  int i = blockIdx.x * blockDim.x + threadIdx.x;     // over 1024*1024
  if (i < HID_ * HID_) {
    int o = i >> 10, k = i & 1023;
    Wt[i] = (bf16)W[k * HID_ + o];
  }
}

__global__ void k_cvt_table(const float* __restrict__ T, bf16* __restrict__ Tt) {
  int i = blockIdx.x * blockDim.x + threadIdx.x;     // over 272*64
  if (i < RELPAD_ * HD_) {
    int rr = i >> 6, d = i & 63;
    Tt[i] = (bf16)(rr < RELROWS_ ? T[rr * HD_ + d] : 0.0f);
  }
}

// ---------------------------------------------------------------------------
// Projection GEMM: Y = X @ Wt^T + b    (X [4096,1024] bf16, Wt [1024,1024] bf16)
// vT==0 : write bf16 at [ (b*NH+h)*S + s ]*64 + d      (head layout)
// vT==1 : write bf16 at [ (b*NH+h)*64 + d ]*S + s      (V transposed)
// Block: 256 thr = 8 waves (2x4); block tile 64(M) x 256(N); wave 32x64.
// ---------------------------------------------------------------------------
__global__ __launch_bounds__(256)
void k_proj(const bf16* __restrict__ X, const bf16* __restrict__ Wt,
            const float* __restrict__ bias, bf16* __restrict__ Yout, int vT)
{
  const int w    = threadIdx.x >> 5;
  const int lane = threadIdx.x & 31;
  const int half = lane >> 4;
  const int r    = lane & 15;
  const int m_base = blockIdx.x * 64  + (w >> 2) * 32;
  const int n_base = blockIdx.y * 256 + (w & 3) * 64;

  v8f acc[2][4] = {};
  for (int kk = 0; kk < HID_; kk += 32) {
    if (kk + 32 < HID_) {
      __builtin_prefetch(X  + (size_t)(m_base + r) * HID_ + kk + 32, 0, 1);
      __builtin_prefetch(Wt + (size_t)(n_base + r) * HID_ + kk + 32, 0, 1);
    }
    v16bf a0 = load_a_frag(X, m_base,      HID_, kk);
    v16bf a1 = load_a_frag(X, m_base + 16, HID_, kk);
    v16bf b0 = load_b_frag(Wt, n_base,      HID_, kk);
    v16bf b1 = load_b_frag(Wt, n_base + 16, HID_, kk);
    v16bf b2 = load_b_frag(Wt, n_base + 32, HID_, kk);
    v16bf b3 = load_b_frag(Wt, n_base + 48, HID_, kk);
    acc[0][0] = wmma_bf16(a0, b0, acc[0][0]);
    acc[0][1] = wmma_bf16(a0, b1, acc[0][1]);
    acc[0][2] = wmma_bf16(a0, b2, acc[0][2]);
    acc[0][3] = wmma_bf16(a0, b3, acc[0][3]);
    acc[1][0] = wmma_bf16(a1, b0, acc[1][0]);
    acc[1][1] = wmma_bf16(a1, b1, acc[1][1]);
    acc[1][2] = wmma_bf16(a1, b2, acc[1][2]);
    acc[1][3] = wmma_bf16(a1, b3, acc[1][3]);
  }

  #pragma unroll
  for (int nt = 0; nt < 4; ++nt) {
    int n = n_base + nt * 16 + r;
    float bv = bias[n];
    int h = n >> 6, d = n & 63;
    #pragma unroll
    for (int mt = 0; mt < 2; ++mt) {
      #pragma unroll
      for (int j = 0; j < 8; ++j) {
        int m = m_base + mt * 16 + j + half * 8;   // global row = b*S + s
        int b = m >> 10, s = m & 1023;
        float v = acc[mt][nt][j] + bv;
        size_t o = vT ? (((size_t)(b * NH_ + h) * HD_ + d) * S_ + s)
                      : (((size_t)(b * NH_ + h) * S_ + s) * HD_ + d);
        Yout[o] = (bf16)v;
      }
    }
  }
}

// ---------------------------------------------------------------------------
// G[bh][m][r] = q[bh][m] . T[r]   ([1024,64] @ [64,272] per head, bf16 WMMA)
// ---------------------------------------------------------------------------
__global__ __launch_bounds__(256)
void k_relpos(const bf16* __restrict__ Q, const bf16* __restrict__ Tt,
              bf16* __restrict__ G)
{
  const int bh   = blockIdx.x;
  const int w    = threadIdx.x >> 5;
  const int lane = threadIdx.x & 31;
  const int half = lane >> 4, r = lane & 15;
  const int m0   = (blockIdx.y * 8 + w) * 16;

  const bf16* q = Q + (size_t)bh * S_ * HD_;
  bf16*       g = G + (size_t)bh * S_ * RELPAD_;

  v16bf a0 = load_a_frag(q, m0, HD_, 0);
  v16bf a1 = load_a_frag(q, m0, HD_, 32);

  for (int nt = 0; nt < 17; ++nt) {
    v16bf b0 = load_b_frag(Tt, nt * 16, HD_, 0);
    v16bf b1 = load_b_frag(Tt, nt * 16, HD_, 32);
    v8f c = {};
    c = wmma_bf16(a0, b0, c);
    c = wmma_bf16(a1, b1, c);
    #pragma unroll
    for (int j = 0; j < 8; ++j) {
      int m = m0 + j + half * 8;
      g[(size_t)m * RELPAD_ + nt * 16 + r] = (bf16)c[j];
    }
  }
}

// ---------------------------------------------------------------------------
// Fused scores + softmax + PV.  One block per (bh, 16-query-row tile).
// 8 waves; wave w owns key columns [w*128, w*128+128).
// ---------------------------------------------------------------------------
__global__ __launch_bounds__(256)
void k_attn_fused(const bf16* __restrict__ Q,  const bf16* __restrict__ Kb,
                  const bf16* __restrict__ SQ, const bf16* __restrict__ SK,
                  const bf16* __restrict__ G,  const bf16* __restrict__ VT,
                  float* __restrict__ Wout, float* __restrict__ Out)
{
  __shared__ bf16  s_p[8][16 * 136];   // per-wave bf16 P slab (pad 136 vs banks)
  __shared__ float s_pv[16 * 64];      // PV cross-wave accumulator
  __shared__ float s_max[8][16];
  __shared__ float s_sum[8][16];

  const int bh   = blockIdx.x;
  const int i0   = blockIdx.y * 16;
  const int w    = threadIdx.x >> 5;
  const int lane = threadIdx.x & 31;
  const int half = lane >> 4;
  const int r    = lane & 15;

  for (int t = threadIdx.x; t < 16 * 64; t += 256) s_pv[t] = 0.0f;

  const bf16* q  = Q  + (size_t)bh * S_ * HD_;
  const bf16* kp = Kb + (size_t)bh * S_ * HD_;
  const bf16* sq = SQ + (size_t)bh * S_ * HD_;
  const bf16* sk = SK + (size_t)bh * S_ * HD_;
  const bf16* g  = G  + (size_t)bh * S_ * RELPAD_;
  const bf16* vt = VT + (size_t)bh * HD_ * S_;

  // ---- Phase A: scores slab 16 x 128 in registers -----------------------
  v16bf aq0 = load_a_frag(q,  i0, HD_, 0);
  v16bf aq1 = load_a_frag(q,  i0, HD_, 32);
  v16bf as0 = load_a_frag(sq, i0, HD_, 0);
  v16bf as1 = load_a_frag(sq, i0, HD_, 32);

  v8f acc[8];
  #pragma unroll
  for (int t = 0; t < 8; ++t) {
    int m0 = w * 128 + t * 16;
    v16bf bk0 = load_b_frag(kp, m0, HD_, 0);
    v16bf bk1 = load_b_frag(kp, m0, HD_, 32);
    v16bf bs0 = load_b_frag(sk, m0, HD_, 0);
    v16bf bs1 = load_b_frag(sk, m0, HD_, 32);
    v8f c = {};
    c = wmma_bf16(aq0, bk0, c);
    c = wmma_bf16(aq1, bk1, c);
    c = wmma_bf16(as0, bs0, c);
    c = wmma_bf16(as1, bs1, c);
    int m = m0 + r;                       // C/D layout: lane -> column m
    #pragma unroll
    for (int j = 0; j < 8; ++j) {
      int i = i0 + j + half * 8;          // C/D layout: VGPR j -> row i
      int rel = m - i;
      rel = rel < -MAXREL_ ? -MAXREL_ : (rel > MAXREL_ ? MAXREL_ : rel);
      float bias = (float)g[(size_t)m * RELPAD_ + rel + MAXREL_];
      c[j] = c[j] * SCALE_ + bias;        // scale*(qk+sqsk) + rel_bias
    }
    acc[t] = c;
  }

  // ---- Phase B: softmax over all 1024 columns ---------------------------
  float pm[8];
  #pragma unroll
  for (int j = 0; j < 8; ++j) {
    float v = acc[0][j];
    #pragma unroll
    for (int t = 1; t < 8; ++t) v = fmaxf(v, acc[t][j]);
    #pragma unroll
    for (int off = 8; off >= 1; off >>= 1) v = fmaxf(v, __shfl_xor(v, off, 32));
    pm[j] = v;
  }
  if (r == 0) {
    #pragma unroll
    for (int j = 0; j < 8; ++j) s_max[w][j + half * 8] = pm[j];
  }
  __syncthreads();

  float rowmax[8];
  #pragma unroll
  for (int j = 0; j < 8; ++j) {
    float v = s_max[0][j + half * 8];
    #pragma unroll
    for (int ww = 1; ww < 8; ++ww) v = fmaxf(v, s_max[ww][j + half * 8]);
    rowmax[j] = v;
  }

  float ps[8];
  #pragma unroll
  for (int j = 0; j < 8; ++j) ps[j] = 0.0f;
  #pragma unroll
  for (int t = 0; t < 8; ++t) {
    #pragma unroll
    for (int j = 0; j < 8; ++j) {
      float e = __expf(acc[t][j] - rowmax[j]);
      acc[t][j] = e;
      ps[j] += e;
    }
  }
  #pragma unroll
  for (int j = 0; j < 8; ++j) {
    float v = ps[j];
    #pragma unroll
    for (int off = 8; off >= 1; off >>= 1) v += __shfl_xor(v, off, 32);
    ps[j] = v;
  }
  if (r == 0) {
    #pragma unroll
    for (int j = 0; j < 8; ++j) s_sum[w][j + half * 8] = ps[j];
  }
  __syncthreads();

  float inv[8];
  #pragma unroll
  for (int j = 0; j < 8; ++j) {
    float v = 0.0f;
    #pragma unroll
    for (int ww = 0; ww < 8; ++ww) v += s_sum[ww][j + half * 8];
    inv[j] = 1.0f / v;
  }

  // normalize; write attn_weights once; stage bf16 P in LDS for PV
  bf16* sp = s_p[w];
  float* wrow = Wout + ((size_t)bh * S_ + i0) * S_;
  #pragma unroll
  for (int t = 0; t < 8; ++t) {
    int mloc = t * 16 + r;
    int m = w * 128 + mloc;
    #pragma unroll
    for (int j = 0; j < 8; ++j) {
      int iloc = j + half * 8;
      float p = acc[t][j] * inv[j];
      wrow[(size_t)iloc * S_ + m] = p;
      sp[iloc * 136 + mloc] = (bf16)p;
    }
  }
  __syncthreads();

  // ---- Phase C: PV partial (16 x 64) over this wave's 128 keys ----------
  v8f oacc[4] = {};
  #pragma unroll
  for (int ks = 0; ks < 4; ++ks) {
    int k0 = ks * 32;
    v16bf a = load_a_frag(sp, 0, 136, k0);                 // ds loads
    #pragma unroll
    for (int nt = 0; nt < 4; ++nt) {
      v16bf b = load_b_frag(vt, nt * 16, S_, w * 128 + k0);
      oacc[nt] = wmma_bf16(a, b, oacc[nt]);
    }
  }
  #pragma unroll
  for (int nt = 0; nt < 4; ++nt) {
    int d = nt * 16 + r;
    #pragma unroll
    for (int j = 0; j < 8; ++j) {
      int iloc = j + half * 8;
      atomicAdd(&s_pv[iloc * 64 + d], oacc[nt][j]);        // ds_add_f32
    }
  }
  __syncthreads();

  for (int t = threadIdx.x; t < 16 * 64; t += 256) {
    int iloc = t >> 6, d = t & 63;
    Out[((size_t)bh * S_ + i0 + iloc) * HD_ + d] = s_pv[t];
  }
}

// ---------------------------------------------------------------------------
// Host launcher
// ---------------------------------------------------------------------------
extern "C" void kernel_launch(void* const* d_in, const int* in_sizes, int n_in,
                              void* d_out, int out_size, void* d_ws, size_t ws_size,
                              hipStream_t stream) {
  (void)in_sizes; (void)n_in; (void)out_size; (void)ws_size;

  const float* hs   = (const float*)d_in[0];
  const float* sph  = (const float*)d_in[1];
  const float* Wq   = (const float*)d_in[2];
  const float* bq   = (const float*)d_in[3];
  const float* Wk   = (const float*)d_in[4];
  const float* bk   = (const float*)d_in[5];
  const float* Wv   = (const float*)d_in[6];
  const float* bv   = (const float*)d_in[7];
  const float* Wqs  = (const float*)d_in[8];
  const float* bqs  = (const float*)d_in[9];
  const float* Wks  = (const float*)d_in[10];
  const float* bks  = (const float*)d_in[11];
  const float* relT = (const float*)d_in[12];

  char* ws = (char*)d_ws;
  size_t off = 0;
  auto alloc = [&](size_t bytes) -> void* {
    void* p = ws + off;
    off = (off + bytes + 255) & ~(size_t)255;
    return p;
  };

  const size_t XN   = (size_t)B_ * S_ * HID_;          // 4,194,304 elems
  const size_t HN   = (size_t)BH_ * S_ * HD_;          // 4,194,304 elems
  bf16* hsb  = (bf16*)alloc(XN * 2);
  bf16* spb  = (bf16*)alloc(XN * 2);
  bf16* wtq  = (bf16*)alloc((size_t)HID_ * HID_ * 2);
  bf16* wtk  = (bf16*)alloc((size_t)HID_ * HID_ * 2);
  bf16* wtv  = (bf16*)alloc((size_t)HID_ * HID_ * 2);
  bf16* wtqs = (bf16*)alloc((size_t)HID_ * HID_ * 2);
  bf16* wtks = (bf16*)alloc((size_t)HID_ * HID_ * 2);
  bf16* tt   = (bf16*)alloc((size_t)RELPAD_ * HD_ * 2);
  bf16* Qb   = (bf16*)alloc(HN * 2);
  bf16* Kbb  = (bf16*)alloc(HN * 2);
  bf16* SQb  = (bf16*)alloc(HN * 2);
  bf16* SKb  = (bf16*)alloc(HN * 2);
  bf16* VTb  = (bf16*)alloc(HN * 2);
  bf16* Gb   = (bf16*)alloc((size_t)BH_ * S_ * RELPAD_ * 2);   // ~35.7 MB

  // 1) conversions
  k_cvt<<<(int)(XN / 256), 256, 0, stream>>>(hs,  hsb, (int)XN);
  k_cvt<<<(int)(XN / 256), 256, 0, stream>>>(sph, spb, (int)XN);
  k_cvt_transpose<<<(HID_ * HID_) / 256, 256, 0, stream>>>(Wq,  wtq);
  k_cvt_transpose<<<(HID_ * HID_) / 256, 256, 0, stream>>>(Wk,  wtk);
  k_cvt_transpose<<<(HID_ * HID_) / 256, 256, 0, stream>>>(Wv,  wtv);
  k_cvt_transpose<<<(HID_ * HID_) / 256, 256, 0, stream>>>(Wqs, wtqs);
  k_cvt_transpose<<<(HID_ * HID_) / 256, 256, 0, stream>>>(Wks, wtks);
  k_cvt_table<<<(RELPAD_ * HD_ + 255) / 256, 256, 0, stream>>>(relT, tt);

  // 2) projections (bf16 WMMA GEMMs)
  dim3 gproj(64, 4);
  k_proj<<<gproj, 256, 0, stream>>>(hsb, wtq,  bq,  Qb,  0);
  k_proj<<<gproj, 256, 0, stream>>>(hsb, wtk,  bk,  Kbb, 0);
  k_proj<<<gproj, 256, 0, stream>>>(hsb, wtv,  bv,  VTb, 1);
  k_proj<<<gproj, 256, 0, stream>>>(spb, wtqs, bqs, SQb, 0);
  k_proj<<<gproj, 256, 0, stream>>>(spb, wtks, bks, SKb, 0);

  // 3) relative-position gather table G (bf16 WMMA GEMM)
  k_relpos<<<dim3(BH_, 8), 256, 0, stream>>>(Qb, tt, Gb);

  // 4) fused scores + softmax + PV
  float* outp = (float*)d_out;                               // [B,NH,S,HD]
  float* wts  = outp + (size_t)B_ * NH_ * S_ * HD_;          // [B,NH,S,S]
  k_attn_fused<<<dim3(BH_, S_ / 16), 256, 0, stream>>>(
      Qb, Kbb, SQb, SKb, Gb, VTb, wts, outp);
}
